// FALayer_20521353740426
// MI455X (gfx1250) — matrix operand order repaired and checked by reading, not compiled, not run on hardware
//
#include <hip/hip_runtime.h>

#define NN 8192
#define HH 256
#define EE 262144
#define LN_EPS 1e-5f

// GEMM tiling
#define KCHUNK 128                    // K elements staged per chunk
#define NCHUNK (NN / KCHUNK)          // 64 chunks
#define ROWH   136                    // padded LDS row stride in halves (272 B = 17*16B)
#define BUFH   (64 * ROWH)            // halves per LDS buffer (64 columns)
#define BUFBYTES (BUFH * 2)           // 17408 B per buffer, x2 = ~34 KB LDS

typedef __bf16 bf16_t;
typedef __attribute__((ext_vector_type(16))) __bf16 v16bf;
typedef __attribute__((ext_vector_type(8)))  __bf16 v8bf;
typedef __attribute__((ext_vector_type(8)))  float  v8f;

union FragBF { v16bf v; v8bf h[2]; };

__device__ __forceinline__ float wave_sum(float v) {
#pragma unroll
  for (int m = 16; m > 0; m >>= 1) v += __shfl_xor(v, m, 32);
  return v;
}

// ---------------------------------------------------------------------------
// Kernel 1: zero the bf16 factor matrix A (134 MB) with wide stores.
// ---------------------------------------------------------------------------
__global__ void zero_ws(uint4* __restrict__ p, int n4) {
  int i = blockIdx.x * blockDim.x + threadIdx.x;
  if (i < n4) p[i] = make_uint4(0u, 0u, 0u, 0u);
}

// ---------------------------------------------------------------------------
// Kernel 2: Bt[n][k] = bf16(inst_feature[k][n])  (256 x 8192 bf16, 4 MB).
// ---------------------------------------------------------------------------
__global__ void feat_to_bf16T(const float* __restrict__ feat,
                              bf16_t* __restrict__ Bt) {
  int i = blockIdx.x * blockDim.x + threadIdx.x;  // 0 .. NN*HH-1
  int k = i & (NN - 1);
  int n = i >> 13;  // NN = 2^13
  Bt[(size_t)n * NN + k] = (bf16_t)feat[(size_t)k * HH + n];
}

// ---------------------------------------------------------------------------
// Kernel 3: one wave32 per edge: LayerNorm(512) + ReLU + gate dot + tanh.
// ---------------------------------------------------------------------------
__global__ void __launch_bounds__(256)
edge_gate(const float* __restrict__ feat,
          const float* __restrict__ nd,
          const int*   __restrict__ pairs,
          const float* __restrict__ gamma,
          const float* __restrict__ beta,
          const float* __restrict__ gw,
          const float* __restrict__ gb,
          float* __restrict__ gOut,
          bf16_t* __restrict__ A) {
  int wid  = threadIdx.x >> 5;
  int lane = threadIdx.x & 31;
  int e    = blockIdx.x * 8 + wid;

  int obj = pairs[2 * e + 0];
  int sub = pairs[2 * e + 1];

  int node = (lane < 16) ? sub : obj;
  int off  = (lane & 15) * 16;
  const float4* src =
      reinterpret_cast<const float4*>(feat + (size_t)node * HH + off);

  float x[16];
#pragma unroll
  for (int j = 0; j < 4; ++j) {
    float4 t = src[j];
    x[4 * j + 0] = t.x; x[4 * j + 1] = t.y;
    x[4 * j + 2] = t.z; x[4 * j + 3] = t.w;
  }

  float s = 0.f, ss = 0.f;
#pragma unroll
  for (int j = 0; j < 16; ++j) { s += x[j]; ss += x[j] * x[j]; }
  s  = wave_sum(s);
  ss = wave_sum(ss);

  float mu   = s * (1.0f / 512.0f);
  float var  = ss * (1.0f / 512.0f) - mu * mu;
  float rstd = rsqrtf(var + LN_EPS);

  int base = lane * 16;
  float dot = 0.f;
#pragma unroll
  for (int j = 0; j < 16; ++j) {
    float xn = (x[j] - mu) * rstd * gamma[base + j] + beta[base + j];
    float h  = fmaxf(xn, 0.f);
    dot += h * gw[base + j];
  }
  dot = wave_sum(dot);

  if (lane == 0) {
    float norm = nd[obj] * nd[sub];
    norm = (norm > 10000.0f) ? 0.f : norm;
    float g = tanhf(dot + gb[0]) * norm;
    gOut[e] = g;
    A[(size_t)sub * NN + obj] = (bf16_t)g;
  }
}

// ---------------------------------------------------------------------------
// Kernel 4: out(8192x256 f32) = A(8192x8192 bf16) @ B(8192x256 bf16).
//
// Block = 8 waves sharing one 64-column B group; each wave owns one 16-row
// M tile and 4 N tiles (4x A-fragment reuse, 4 f32 accumulators). B chunks
// (64x128 bf16) are staged into double-buffered LDS with
// GLOBAL_LOAD_ASYNC_TO_LDS_B128 (ASYNCcnt). Compiler-visible dummy ds_stores
// into shB defeat the "LDS never written -> loads are undef" fold (the async
// writes happen inside asm the compiler cannot model); barriers order the
// real data flow. A fragments for the whole chunk are preloaded so global
// load latency overlaps WMMA issue.
// ---------------------------------------------------------------------------
__global__ void __launch_bounds__(256)
wmma_gemm(const bf16_t* __restrict__ A,
          const bf16_t* __restrict__ Bt,
          float* __restrict__ out) {
  __shared__ __align__(16) bf16_t shB[2 * BUFH];

  int tid  = threadIdx.x;
  int wid  = tid >> 5;
  int lane = tid & 31;

  int ntg = blockIdx.x & 3;        // N group: 64 columns
  int mtg = blockIdx.x >> 2;       // 0..63
  int mt  = mtg * 8 + wid;         // 0..511: this wave's 16-row M tile
  int n0  = ntg * 64;

  int mrow = mt * 16 + (lane & 15);
  const bf16_t* Arow = A + (size_t)mrow * NN;
  int aoff = (lane < 16) ? 0 : 8;   // A-frag K base per lane half
  int boff = (lane < 16) ? 0 : 8;   // B-frag K base per lane half

  v8f acc[4] = {};

  // Compiler-visible stores into shB at dynamic addresses: later ds_loads
  // may-alias these, so they can neither be undef-folded nor forwarded.
  shB[tid]        = (bf16_t)0.0f;
  shB[BUFH + tid] = (bf16_t)0.0f;
  __syncthreads();

  // --- async stage of one B chunk: 1024 x 16B segments, 4 per thread -------
  auto issue_chunk = [&](int k0, uint32_t bufBaseBytes) {
#pragma unroll
    for (int i = 0; i < 4; ++i) {
      int s   = i * 256 + tid;      // 0..1023, lanes contiguous per pass
      int row = s >> 4;             // 0..63  (local column index)
      int seg = s & 15;             // 16B segment within the 256B row
      uint32_t ldsoff = bufBaseBytes + (uint32_t)(row * ROWH + seg * 8) * 2u;
      const bf16_t* g = Bt + (size_t)(n0 + row) * NN + k0 + seg * 8;
      asm volatile("global_load_async_to_lds_b128 %0, %1, off"
                   :: "v"(ldsoff), "v"(g)
                   : "memory");
    }
  };

  issue_chunk(0, 0);

  for (int c = 0; c < NCHUNK; ++c) {
    uint32_t curBase = (c & 1) ? (uint32_t)BUFBYTES : 0u;
    if (c + 1 < NCHUNK) {
      issue_chunk((c + 1) * KCHUNK, (c & 1) ? 0u : (uint32_t)BUFBYTES);
      // 4 new loads outstanding; wait until only those remain -> chunk c done
      asm volatile("s_wait_asynccnt 4" ::: "memory");
    } else {
      asm volatile("s_wait_asynccnt 0" ::: "memory");
    }
    __syncthreads();  // chunk c visible to all waves

    const bf16_t* shBuf = shB + (curBase >> 1);

    // Preload all 4 A fragments of this chunk (one clause of 8 b128 loads).
    FragBF afr[4];
#pragma unroll
    for (int ks = 0; ks < 4; ++ks) {
      int kabs = c * KCHUNK + ks * 32;
      afr[ks].h[0] = *reinterpret_cast<const v8bf*>(Arow + kabs + aoff);
      afr[ks].h[1] = *reinterpret_cast<const v8bf*>(Arow + kabs + aoff + 16);
    }
    __builtin_prefetch(Arow + c * KCHUNK + 2 * KCHUNK, 0, 1);

#pragma unroll
    for (int ks = 0; ks < 4; ++ks) {             // 4 x K=32 steps per chunk
#pragma unroll
      for (int t = 0; t < 4; ++t) {              // 4 N tiles share this A frag
        int coll = t * 16 + (lane & 15);
        const bf16_t* bp = shBuf + coll * ROWH + ks * 32 + boff;
        FragBF b;
        b.h[0] = *reinterpret_cast<const v8bf*>(bp);       // K..K+7
        b.h[1] = *reinterpret_cast<const v8bf*>(bp + 16);  // K+16..K+23
        acc[t] = __builtin_amdgcn_wmma_f32_16x16x32_bf16(
            false, afr[ks].v, false, b.v, (short)0, acc[t], false, false);
      }
    }
    __syncthreads();  // everyone done with chunk c before it is overwritten
  }

  // C/D layout: VGPR r, lanes 0-15 -> M = 16*mt + r, lanes 16-31 -> M + 8.
  int mbase = mt * 16 + ((lane < 16) ? 0 : 8);
#pragma unroll
  for (int t = 0; t < 4; ++t) {
    int nout = n0 + t * 16 + (lane & 15);
#pragma unroll
    for (int r = 0; r < 8; ++r) {
      out[(size_t)(mbase + r) * HH + nout] = acc[t][r];
    }
  }
}

// ---------------------------------------------------------------------------
// Launch: zero A -> build Bt -> edge gate/scatter -> WMMA GEMM.
// ---------------------------------------------------------------------------
extern "C" void kernel_launch(void* const* d_in, const int* in_sizes, int n_in,
                              void* d_out, int out_size, void* d_ws,
                              size_t ws_size, hipStream_t stream) {
  const float* feat  = (const float*)d_in[0];   // (N,H) f32
  const float* nd    = (const float*)d_in[1];   // (N,)  f32
  const int*   pairs = (const int*)  d_in[2];   // (E,2) int
  const float* gamma = (const float*)d_in[3];   // (512,)
  const float* beta  = (const float*)d_in[4];   // (512,)
  const float* gw    = (const float*)d_in[5];   // (512,1)
  const float* gb    = (const float*)d_in[6];   // (1,)

  float* aggOut = (float*)d_out;                // (N,H) f32
  float* gOut   = aggOut + (size_t)NN * HH;     // (E,)  f32

  bf16_t* A  = (bf16_t*)d_ws;                                             // 134 MB
  bf16_t* Bt = (bf16_t*)((char*)d_ws + (size_t)NN * NN * sizeof(bf16_t)); // 4 MB

  int n4 = (int)(((size_t)NN * NN * sizeof(bf16_t)) / 16);
  zero_ws<<<(n4 + 255) / 256, 256, 0, stream>>>((uint4*)A, n4);

  feat_to_bf16T<<<(NN * HH + 255) / 256, 256, 0, stream>>>(feat, Bt);

  edge_gate<<<EE / 8, 256, 0, stream>>>(feat, nd, pairs, gamma, beta, gw, gb,
                                        gOut, A);

  // 64 M-groups x 4 N-groups = 256 blocks of 8 waves
  wmma_gemm<<<256, 256, 0, stream>>>(A, Bt, aggOut);
}
